// MultiHeadRelativeAttention_65403761983780
// MI455X (gfx1250) — matrix operand order repaired and checked by reading, not compiled, not run on hardware
//
#include <hip/hip_runtime.h>
#include <hip/hip_bf16.h>
#include <math.h>

typedef __attribute__((ext_vector_type(16))) __bf16 bf16x16;
typedef __attribute__((ext_vector_type(8)))  __bf16 bf16x8;
typedef __attribute__((ext_vector_type(8)))  float  f32x8;
typedef __attribute__((ext_vector_type(4)))  float  f32x4;

#define D_MODEL 1024
#define N_HEADS 16
#define D_K     64
#define SEQ     2048
#define BATCH   2
#define M_TOK   (BATCH*SEQ)   // 4096

// ---------------------------------------------------------------------------
// helpers
// ---------------------------------------------------------------------------
static __device__ __forceinline__ f32x8 wmma_bf16(bf16x16 a, bf16x16 b, f32x8 c) {
  return __builtin_amdgcn_wmma_f32_16x16x32_bf16(false, a, false, b, (short)0, c,
                                                 false, false);
}

// 16x32 bf16 A-fragment from row-major [16 x ld] source.
// Layout: lanes 0-15 row M=lane, K=0-15;
//         lanes 16-31 row M=lane-16, K=16-31.
static __device__ __forceinline__ bf16x16 load_a_frag(const __bf16* base, int ld) {
  int lane = threadIdx.x & 31;
  int r = lane & 15, h = lane >> 4;
  const __bf16* p = base + (size_t)r * ld + h * 16;
  bf16x8 lo = *(const bf16x8*)p;
  bf16x8 hi = *(const bf16x8*)(p + 8);
  bf16x16 o;
#pragma unroll
  for (int i = 0; i < 8; i++) { o[i] = lo[i]; o[i + 8] = hi[i]; }
  return o;
}

// 32x16 bf16 B-fragment where logical B[k][n] = src[n][k], src row-major [16 x ld].
// Layout: lane holds column n=lane&15; lanes 0-15 K=0-15, lanes 16-31 K=16-31.
static __device__ __forceinline__ bf16x16 load_b_frag(const __bf16* src, int ld) {
  int lane = threadIdx.x & 31;
  int c = lane & 15, h = lane >> 4;
  return *(const bf16x16*)(src + (size_t)c * ld + h * 16);
}

// ---------------------------------------------------------------------------
// f32 -> bf16 conversion (8 elems / thread)
// ---------------------------------------------------------------------------
__global__ void cvt_f32_to_bf16(const float* __restrict__ in,
                                __bf16* __restrict__ out, int n8) {
  int i = blockIdx.x * blockDim.x + threadIdx.x;
  if (i >= n8) return;
  const f32x4* p = (const f32x4*)(in + (size_t)i * 8);
  f32x4 a = p[0], b = p[1];
  bf16x8 o;
#pragma unroll
  for (int j = 0; j < 4; j++) { o[j] = (__bf16)a[j]; o[4 + j] = (__bf16)b[j]; }
  *(bf16x8*)(out + (size_t)i * 8) = o;
}

// ---------------------------------------------------------------------------
// relative-position bias: bias[h][d], d = (key - query) + 2047, d in [0,4094]
// ---------------------------------------------------------------------------
__global__ void build_bias(const float* __restrict__ table, float* __restrict__ bias) {
  int idx = blockIdx.x * blockDim.x + threadIdx.x;  // h*4096 + d
  int h = idx >> 12, d = idx & 4095;
  if (h >= N_HEADS) return;
  float val = 0.f;
  if (d < 4095) {
    int rp = d - 2047;      // key - query  (memory_pos - context_pos)
    int n = -rp;
    int ret = 0;
    if (n < 0) { ret = 16; n = -n; }   // bidirectional: +num_buckets/2
    int b;
    if (n < 8) {
      b = n;
    } else {
      // 8 + log(n/8)/log(16) * 8, clamped to 15
      b = 8 + (int)(__logf((float)n * 0.125f) * (8.0f / __logf(16.0f)));
      if (b > 15) b = 15;
    }
    val = table[(ret + b) * N_HEADS + h];
  }
  bias[idx] = val;
}

// ---------------------------------------------------------------------------
// GEMM: C[M,N] = A[M,K] * W[N,K]^T   (bf16 in, f32 acc; bf16 or f32 out)
// block = 256 thr (8 waves, 2Mx4N), wave tile 64Mx32N, block tile 128x128.
// Fragment registers are double-buffered: next K-chunk's 12 loads are issued
// before the current chunk's 8 WMMAs so global latency hides under matrix ops.
// ---------------------------------------------------------------------------
template <bool F32OUT>
__global__ __launch_bounds__(256)
void gemm_nt_bf16(const __bf16* __restrict__ A, const __bf16* __restrict__ W,
                  void* __restrict__ Cout, float scale) {
  const int K = D_MODEL, N = D_MODEL;
  int w = threadIdx.x >> 5;
  int lane = threadIdx.x & 31;
  int wm = w >> 2;               // 0..1
  int wn = w & 3;                // 0..3
  int m0 = blockIdx.x * 128 + wm * 64;
  int n0 = blockIdx.y * 128 + wn * 32;

  const __bf16* Abase = A + (size_t)m0 * K;
  const __bf16* Wbase = W + (size_t)n0 * K;

  bf16x16 a[4], bb[2];
#pragma unroll
  for (int i = 0; i < 4; i++) a[i] = load_a_frag(Abase + (size_t)(16 * i) * K, K);
#pragma unroll
  for (int j = 0; j < 2; j++) bb[j] = load_b_frag(Wbase + (size_t)(16 * j) * K, K);

  f32x8 acc[4][2] = {};
  for (int kk = 0; kk < K; kk += 32) {
    bool more = (kk + 32 < K);
    bf16x16 an[4], bn[2];
    if (more) {
#pragma unroll
      for (int i = 0; i < 4; i++)
        an[i] = load_a_frag(Abase + (size_t)(16 * i) * K + kk + 32, K);
#pragma unroll
      for (int j = 0; j < 2; j++)
        bn[j] = load_b_frag(Wbase + (size_t)(16 * j) * K + kk + 32, K);
    }
#pragma unroll
    for (int i = 0; i < 4; i++)
#pragma unroll
      for (int j = 0; j < 2; j++)
        acc[i][j] = wmma_bf16(a[i], bb[j], acc[i][j]);
    if (more) {
#pragma unroll
      for (int i = 0; i < 4; i++) a[i] = an[i];
#pragma unroll
      for (int j = 0; j < 2; j++) bb[j] = bn[j];
    }
  }

  int c = lane & 15, hf = lane >> 4;
#pragma unroll
  for (int ti = 0; ti < 4; ti++)
#pragma unroll
    for (int tj = 0; tj < 2; tj++) {
#pragma unroll
      for (int i = 0; i < 8; i++) {
        int row = m0 + ti * 16 + hf * 8 + i;
        int col = n0 + tj * 16 + c;
        float v = acc[ti][tj][i] * scale;
        if (F32OUT) ((float*)Cout)[(size_t)row * N + col] = v;
        else        ((__bf16*)Cout)[(size_t)row * N + col] = (__bf16)v;
      }
    }
}

// ---------------------------------------------------------------------------
// Flash attention, one (b,h) x 128-query tile per block; 8 waves x 16 queries.
// Scores computed transposed: S^T[key,q] = K_tile x Q^T, so per-query softmax
// reductions need only one shfl_xor(16). Output: O^T = V^T x P^T.
// K/V/bias tiles are LDS double-buffered (ping-pong): one barrier per tile,
// tile t+1 global loads overlap tile t's WMMAs.
// ---------------------------------------------------------------------------
__global__ __launch_bounds__(256)
void flash_attn(const __bf16* __restrict__ Q, const __bf16* __restrict__ Kmat,
                const __bf16* __restrict__ V, const float* __restrict__ bias,
                __bf16* __restrict__ O) {
  __shared__ __attribute__((aligned(64))) __bf16 Klds[2][32 * 64];     // [key][dk]
  __shared__ __attribute__((aligned(64))) __bf16 Vlds[2][64 * 40];     // [dk][key] pad 40
  __shared__ __attribute__((aligned(64))) float  biasS[2][160];        // bias window
  __shared__ __attribute__((aligned(64))) __bf16 Plds[8][16 * 32];     // per-wave [q][key]

  int b = blockIdx.y >> 4;
  int h = blockIdx.y & 15;
  int w = threadIdx.x >> 5;
  int lane = threadIdx.x & 31;
  int c = lane & 15, hh = lane >> 4;
  int q0blk = blockIdx.x * 128;
  int q0 = q0blk + w * 16;
  size_t tok0 = (size_t)b * SEQ;

  const float* biasH = bias + (size_t)h * 4096;

  // Q^T B-fragments (dk 0-31 and 32-63), held in registers for all key tiles
  const __bf16* Qbase = Q + (tok0 + q0) * D_MODEL + h * D_K;
  bf16x16 bq0 = load_b_frag(Qbase + 0, D_MODEL);
  bf16x16 bq1 = load_b_frag(Qbase + 32, D_MODEL);

  f32x8 acco[4] = {};                    // O^T, dk tiles 0..3
  float m = -INFINITY, l = 0.f;

  int tid = threadIdx.x;
  int skey = tid >> 3;                   // 0..31 key within tile
  int sd8 = (tid & 7) * 8;               // dk octet
  // tile-invariant offset into biasS: d - dmin = sub*16 + hh*8 + i + bbase
  int bbase = 127 - (w * 16 + c);

  const int NT = SEQ / 32;               // 64 key tiles

  bf16x8 kreg, vreg;
  float breg = 0.f;
  auto load_tile = [&](int t) {
    size_t rowoff = (tok0 + t * 32 + skey) * D_MODEL + h * D_K + sd8;
    kreg = *(const bf16x8*)(Kmat + rowoff);
    vreg = *(const bf16x8*)(V + rowoff);
    if (tid < 160) breg = biasH[t * 32 - q0blk + 1920 + tid];
  };
  auto store_tile = [&](int buf) {
    *(bf16x8*)&Klds[buf][skey * 64 + sd8] = kreg;
#pragma unroll
    for (int i = 0; i < 8; i++) Vlds[buf][(sd8 + i) * 40 + skey] = vreg[i];
    if (tid < 160) biasS[buf][tid] = breg;
  };

  load_tile(0);
  store_tile(0);
  __syncthreads();

  for (int t = 0; t < NT; t++) {
    int cur = t & 1;
    bool more = (t + 1 < NT);
    if (more) load_tile(t + 1);          // overlaps with compute below

    // S^T tiles: [16 keys x 16 q] x 2 (keys 0-15 and 16-31)
    f32x8 st[2] = {};
#pragma unroll
    for (int sub = 0; sub < 2; sub++) {
      bf16x16 ak0 = load_a_frag(&Klds[cur][sub * 16 * 64 + 0], 64);
      bf16x16 ak1 = load_a_frag(&Klds[cur][sub * 16 * 64 + 32], 64);
      st[sub] = wmma_bf16(ak0, bq0, st[sub]);
      st[sub] = wmma_bf16(ak1, bq1, st[sub]);
    }

    // relative-position bias from the staged LDS window
#pragma unroll
    for (int sub = 0; sub < 2; sub++) {
      int ib = bbase + sub * 16 + hh * 8;
#pragma unroll
      for (int i = 0; i < 8; i++) st[sub][i] += biasS[cur][ib + i];
    }

    // online softmax over keys (per query column = lane pair {c, c+16})
    float tmax = st[0][0];
#pragma unroll
    for (int i = 1; i < 8; i++) tmax = fmaxf(tmax, st[0][i]);
#pragma unroll
    for (int i = 0; i < 8; i++) tmax = fmaxf(tmax, st[1][i]);
    tmax = fmaxf(tmax, __shfl_xor(tmax, 16, 32));
    float newm = fmaxf(m, tmax);
    float corr = __expf(m - newm);

    float rs = 0.f;
    bf16x8 p[2];
#pragma unroll
    for (int sub = 0; sub < 2; sub++)
#pragma unroll
      for (int i = 0; i < 8; i++) {
        float e = __expf(st[sub][i] - newm);
        rs += e;
        p[sub][i] = (__bf16)e;
      }
    l = l * corr + rs + __shfl_xor(rs, 16, 32);
    m = newm;
#pragma unroll
    for (int dt = 0; dt < 4; dt++)
#pragma unroll
      for (int i = 0; i < 8; i++) acco[dt][i] *= corr;

    // stash P^T through per-wave LDS to relayout C-frag -> B-frag
    *(bf16x8*)&Plds[w][c * 32 + 0 * 16 + hh * 8] = p[0];
    *(bf16x8*)&Plds[w][c * 32 + 1 * 16 + hh * 8] = p[1];

    // O^T += V^T x P^T   (K-dim = 32 keys, one wmma per dk tile)
    bf16x16 bp = load_b_frag(&Plds[w][0], 32);
#pragma unroll
    for (int dt = 0; dt < 4; dt++) {
      bf16x16 av = load_a_frag(&Vlds[cur][dt * 16 * 40], 40);
      acco[dt] = wmma_bf16(av, bp, acco[dt]);
    }

    if (more) store_tile(cur ^ 1);       // fill the other buffer
    __syncthreads();
  }

  // normalize and store: lane holds query column c, contiguous dk per register
  float invl = 1.0f / l;
  __bf16* Obase = O + (tok0 + q0 + c) * D_MODEL + h * D_K;
#pragma unroll
  for (int dt = 0; dt < 4; dt++) {
    bf16x8 o;
#pragma unroll
    for (int i = 0; i < 8; i++) o[i] = (__bf16)(acco[dt][i] * invl);
    *(bf16x8*)&Obase[dt * 16 + hh * 8] = o;
  }
}

// ---------------------------------------------------------------------------
// launch
// ---------------------------------------------------------------------------
extern "C" void kernel_launch(void* const* d_in, const int* in_sizes, int n_in,
                              void* d_out, int out_size, void* d_ws, size_t ws_size,
                              hipStream_t stream) {
  const float* x   = (const float*)d_in[0];
  const float* wq  = (const float*)d_in[1];
  const float* wk  = (const float*)d_in[2];
  const float* wv  = (const float*)d_in[3];
  const float* wo  = (const float*)d_in[4];
  const float* tbl = (const float*)d_in[5];

  char* ws = (char*)d_ws;
  size_t off = 0;
  auto alloc = [&](size_t bytes) {
    void* p = ws + off;
    off += (bytes + 255) & ~(size_t)255;
    return p;
  };

  __bf16* xb  = (__bf16*)alloc((size_t)M_TOK * D_MODEL * 2);
  __bf16* wqb = (__bf16*)alloc((size_t)D_MODEL * D_MODEL * 2);
  __bf16* wkb = (__bf16*)alloc((size_t)D_MODEL * D_MODEL * 2);
  __bf16* wvb = (__bf16*)alloc((size_t)D_MODEL * D_MODEL * 2);
  __bf16* wob = (__bf16*)alloc((size_t)D_MODEL * D_MODEL * 2);
  __bf16* qb  = (__bf16*)alloc((size_t)M_TOK * D_MODEL * 2);
  __bf16* kb  = (__bf16*)alloc((size_t)M_TOK * D_MODEL * 2);
  __bf16* vb  = (__bf16*)alloc((size_t)M_TOK * D_MODEL * 2);
  __bf16* ao  = (__bf16*)alloc((size_t)M_TOK * D_MODEL * 2);
  float*  bias = (float*)alloc((size_t)N_HEADS * 4096 * 4);

  int n8x = M_TOK * D_MODEL / 8;
  cvt_f32_to_bf16<<<(n8x + 255) / 256, 256, 0, stream>>>(x, xb, n8x);
  int n8w = D_MODEL * D_MODEL / 8;
  cvt_f32_to_bf16<<<(n8w + 255) / 256, 256, 0, stream>>>(wq, wqb, n8w);
  cvt_f32_to_bf16<<<(n8w + 255) / 256, 256, 0, stream>>>(wk, wkb, n8w);
  cvt_f32_to_bf16<<<(n8w + 255) / 256, 256, 0, stream>>>(wv, wvb, n8w);
  cvt_f32_to_bf16<<<(n8w + 255) / 256, 256, 0, stream>>>(wo, wob, n8w);

  build_bias<<<(N_HEADS * 4096) / 256, 256, 0, stream>>>(tbl, bias);

  dim3 ggrid(M_TOK / 128, D_MODEL / 128);
  // scores scale 1/sqrt(64) folded into Q
  gemm_nt_bf16<false><<<ggrid, 256, 0, stream>>>(xb, wqb, qb, 0.125f);
  gemm_nt_bf16<false><<<ggrid, 256, 0, stream>>>(xb, wkb, kb, 1.0f);
  gemm_nt_bf16<false><<<ggrid, 256, 0, stream>>>(xb, wvb, vb, 1.0f);

  flash_attn<<<dim3(SEQ / 128, BATCH * N_HEADS), 256, 0, stream>>>(qb, kb, vb,
                                                                   bias, ao);

  gemm_nt_bf16<true><<<ggrid, 256, 0, stream>>>(ao, wob, d_out, 1.0f);
}
